// MyModel_42064909697472
// MI455X (gfx1250) — matrix operand order repaired and checked by reading, not compiled, not run on hardware
//
#include <hip/hip_runtime.h>
#include <math.h>

// ---------------- CDNA5 WMMA types ----------------
typedef __attribute__((ext_vector_type(16))) __bf16 v16bf;
typedef __attribute__((ext_vector_type(8)))  __bf16 v8bf;
typedef __attribute__((ext_vector_type(4)))  __bf16 v4bf;
typedef __attribute__((ext_vector_type(8)))  float  v8f;
typedef __attribute__((ext_vector_type(4)))  unsigned int v4u;
typedef __attribute__((ext_vector_type(8)))  int v8i;
typedef __attribute__((ext_vector_type(4)))  int v4i;

static constexpr int kE = 512, kH = 1024, kV = 32000, kB = 16, kS = 256;
static constexpr int kR = kB * kS;        // 4096 (b*S+s) rows
static constexpr int kParts = kV / 256;   // 125 vocab column-groups

#if defined(__AMDGCN__) && __has_builtin(__builtin_amdgcn_tensor_load_to_lds) && \
    __has_builtin(__builtin_amdgcn_s_wait_tensorcnt)
#define USE_TDM 1
#else
#define USE_TDM 0
#endif

// ---- WMMA operand loaders (contiguous k = 16*(lane/16) + s in both A and B, wave32) ----
__device__ __forceinline__ v16bf wmma_load_a(const __bf16* __restrict__ A, int lda,
                                             int mbase, int k0, int lane) {
  const int m  = mbase + (lane & 15);
  const int kb = (lane < 16) ? 0 : 16;
  const __bf16* p = A + (size_t)m * lda + k0 + kb;
  v8bf lo = *(const v8bf*)p;
  v8bf hi = *(const v8bf*)(p + 8);
  v16bf a;
#pragma unroll
  for (int i = 0; i < 8; ++i) { a[i] = lo[i]; a[i + 8] = hi[i]; }
  return a;
}

__device__ __forceinline__ v16bf wmma_load_b(const __bf16* __restrict__ W, int ldw,
                                             int nbase, int k0, int lane) {
  const int n  = nbase + (lane & 15);
  const int kb = (lane < 16) ? 0 : 16;
  return *(const v16bf*)(W + (size_t)n * ldw + k0 + kb);
}

__device__ __forceinline__ v8f wmma_bf16(v16bf a, v16bf b, v8f c) {
  return __builtin_amdgcn_wmma_f32_16x16x32_bf16(false, a, false, b, (short)0, c,
                                                 false, false);
}

#if USE_TDM
// 2D TDM load: tile (tile_d0 x tile_d1) of a (tensor_d0 x tensor_d1) bf16
// tensor (row stride = stride0 elements) -> contiguous LDS at lds_off.
// D# packing per CDNA5 ISA 8.3/8.4 (group0 128b, group1 256b, groups 2/3 zero).
__device__ __forceinline__ void tdm_load_2d_bf16(unsigned int lds_off,
    const void* gaddr, unsigned int tensor_d0, unsigned int tensor_d1,
    unsigned int tile_d0, unsigned int tile_d1, unsigned int stride0) {
  const unsigned long long ga = (unsigned long long)gaddr;
  v4u g0;
  g0[0] = 1u;                                            // count=1, user D#
  g0[1] = lds_off;                                       // lds_addr
  g0[2] = (unsigned int)ga;                              // global_addr[31:0]
  g0[3] = (unsigned int)((ga >> 32) & 0x1FFFFFFu) | (2u << 30);  // [56:32]|type=2
  v8i g1;
  g1[0] = (int)(1u << 16);                               // mask=0, data_size=1 (2B)
  g1[1] = (int)((tensor_d0 & 0xFFFFu) << 16);            // tensor_dim0[15:0]
  g1[2] = (int)(((tensor_d0 >> 16) & 0xFFFFu) | ((tensor_d1 & 0xFFFFu) << 16));
  g1[3] = (int)(((tensor_d1 >> 16) & 0xFFFFu) | ((tile_d0 & 0xFFFFu) << 16));
  g1[4] = (int)(tile_d1 & 0xFFFFu);                      // tile_dim1, tile_dim2=0
  g1[5] = (int)stride0;                                  // dim0_stride[31:0]
  g1[6] = 0;                                             // dim0_stride[47:32], dim1_stride lo
  g1[7] = 0;
  const v4i gz = {0, 0, 0, 0};
#if defined(__clang_major__) && (__clang_major__ >= 23)
  const v8i gz8 = {0, 0, 0, 0, 0, 0, 0, 0};
  __builtin_amdgcn_tensor_load_to_lds(g0, g1, gz, gz, gz8, 0);
#else
  __builtin_amdgcn_tensor_load_to_lds(g0, g1, gz, gz, 0);
#endif
}
#endif

// ---------------- fp32 -> bf16 conversion ----------------
__global__ __launch_bounds__(256) void cvt_bf16(const float* __restrict__ src,
                                                __bf16* __restrict__ dst, int n) {
  const int i = (blockIdx.x * 256 + threadIdx.x) * 4;
  if (i + 3 < n) {
    const float4 v = *(const float4*)(src + i);
    v4bf o;
    o[0] = (__bf16)v.x; o[1] = (__bf16)v.y;
    o[2] = (__bf16)v.z; o[3] = (__bf16)v.w;
    *(v4bf*)(dst + i) = o;
  } else if (i < n) {
    for (int j = i; j < n; ++j) dst[j] = (__bf16)src[j];
  }
}

// ---------------- xp = x @ W_ih^T + (b_ih + b_hh) ----------------
__global__ __launch_bounds__(128) void gemm_xp(const __bf16* __restrict__ X,
    const __bf16* __restrict__ Wih, const float* __restrict__ bih,
    const float* __restrict__ bhh, float* __restrict__ XP) {
  const int lane = threadIdx.x & 31, wave = threadIdx.x >> 5;
  const int mbase = blockIdx.x * 16;
  const int nbase = (blockIdx.y * 4 + wave) * 16;
  v8f acc = {};
#pragma unroll
  for (int k0 = 0; k0 < kE; k0 += 32) {
    v16bf a = wmma_load_a(X, kE, mbase, k0, lane);
    v16bf b = wmma_load_b(Wih, kE, nbase, k0, lane);
    acc = wmma_bf16(a, b, acc);
  }
  const int col  = nbase + (lane & 15);
  const float bv = bih[col] + bhh[col];
  const int roff = (lane < 16) ? 0 : 8;
#pragma unroll
  for (int r = 0; r < 8; ++r)
    XP[(size_t)(mbase + r + roff) * kH + col] = acc[r] + bv;
}

// ------------ one recurrence step: h = tanh(xp_t + h @ W_hh^T) ------------
__global__ __launch_bounds__(128) void rnn_step(const __bf16* __restrict__ hprev,
    const __bf16* __restrict__ Whh, const float* __restrict__ XP, int t,
    float* __restrict__ HSf, __bf16* __restrict__ HSb, __bf16* __restrict__ hnext) {
  __shared__ v8f red[4][32];
  const int lane = threadIdx.x & 31, wave = threadIdx.x >> 5;
  const int nbase = blockIdx.x * 16;
  const int kbeg = wave * (kH / 4);
  v8f acc = {};
#pragma unroll
  for (int kk = 0; kk < kH / 4; kk += 32) {
    const int k0 = kbeg + kk;
    v16bf a = wmma_load_a(hprev, kH, 0, k0, lane);
    v16bf b = wmma_load_b(Whh, kH, nbase, k0, lane);
    acc = wmma_bf16(a, b, acc);
  }
  red[wave][lane] = acc;
  __syncthreads();
  if (wave == 0) {
    v8f c = red[0][lane];
#pragma unroll
    for (int w = 1; w < 4; ++w) c += red[w][lane];
    const int col  = nbase + (lane & 15);
    const int roff = (lane < 16) ? 0 : 8;
#pragma unroll
    for (int r = 0; r < 8; ++r) {
      const int bidx = r + roff;
      const size_t row = (size_t)bidx * kS + t;
      const float h = tanhf(c[r] + XP[row * kH + col]);
      HSf[row * kH + col] = h;
      const __bf16 hb = (__bf16)h;
      HSb[row * kH + col] = hb;
      hnext[(size_t)bidx * kH + col] = hb;
    }
  }
}

// ------- fused vocab GEMM tile + online logsumexp partials -------
// grid (R/64, V/256), block 256 (8 waves). Block tile: 64 rows x 256 cols.
// Wave w: M-tiles {2*(w&1), 2*(w&1)+1}, col group (w>>1)*64 -> 8 accumulators.
// B K-slices (256 rows x 32 K = 16 KB) are TDM double-buffered in LDS.
__global__ __launch_bounds__(256) void vocab_lse(const __bf16* __restrict__ HSb,
    const __bf16* __restrict__ VWb, const float* __restrict__ Vb,
    float2* __restrict__ part) {
  __shared__ float tile[64][260];   // logit spill; first 32 KB aliased as B stage
  __shared__ float pmax[64][4];
  __shared__ float psum[64][4];
  const int lane = threadIdx.x & 31, wave = threadIdx.x >> 5;
  const int rowbase = blockIdx.x * 64;
  const int colbase = blockIdx.y * 256;
  const int mgrp = wave & 1;        // rows [mgrp*32, +32)
  const int ngrp = wave >> 1;       // cols [ngrp*64, +64)
  const int mb0 = rowbase + mgrp * 32;
  v8f acc[2][4] = {{{}, {}, {}, {}}, {{}, {}, {}, {}}};

#if USE_TDM
  __bf16* bst = (__bf16*)&tile[0][0];
  __bf16* bufs[2] = {bst, bst + 8192};         // 2 x 16 KB staging buffers
  const char* gbase = (const char*)VWb + (size_t)colbase * kH * 2;
  if (wave == 0)                                // prime slice 0
    tdm_load_2d_bf16((unsigned int)(uintptr_t)bufs[0], gbase,
                     kH, kV, 32, 256, kH);
  int cur = 0;
  for (int kit = 0; kit < kH / 32; ++kit) {
    if (wave == 0) __builtin_amdgcn_s_wait_tensorcnt(0);   // slice kit landed
    __syncthreads();                                       // publish to block
    if (wave == 0 && kit + 1 < kH / 32)                    // prefetch slice kit+1
      tdm_load_2d_bf16((unsigned int)(uintptr_t)bufs[cur ^ 1],
                       gbase + (size_t)(kit + 1) * 64, kH, kV, 32, 256, kH);
    const int k0 = kit * 32;
    const __bf16* bb = bufs[cur];
    v16bf a0 = wmma_load_a(HSb, kH, mb0, k0, lane);
    v16bf a1 = wmma_load_a(HSb, kH, mb0 + 16, k0, lane);
    const int kb = (lane < 16) ? 0 : 16;
#pragma unroll
    for (int j = 0; j < 4; ++j) {
      const int cl = ngrp * 64 + j * 16 + (lane & 15);     // local B column
      v16bf b = *(const v16bf*)(bb + cl * 32 + kb);        // ds_load from stage
      acc[0][j] = wmma_bf16(a0, b, acc[0][j]);
      acc[1][j] = wmma_bf16(a1, b, acc[1][j]);
    }
    __syncthreads();          // all reads of bufs[cur] done before reuse
    cur ^= 1;
  }
#else
  for (int k0 = 0; k0 < kH; k0 += 32) {
    v16bf a0 = wmma_load_a(HSb, kH, mb0, k0, lane);
    v16bf a1 = wmma_load_a(HSb, kH, mb0 + 16, k0, lane);
#pragma unroll
    for (int j = 0; j < 4; ++j) {
      const int nb = colbase + ngrp * 64 + j * 16;
      if (k0 + 32 < kH)
        __builtin_prefetch(VWb + (size_t)(nb + (lane & 15)) * kH + k0 + 32, 0, 1);
      v16bf b = wmma_load_b(VWb, kH, nb, k0, lane);
      acc[0][j] = wmma_bf16(a0, b, acc[0][j]);
      acc[1][j] = wmma_bf16(a1, b, acc[1][j]);
    }
  }
  __syncthreads();
#endif

  // spill logits (+Vb) into LDS tile
  const int roff = (lane < 16) ? 0 : 8;
#pragma unroll
  for (int mt = 0; mt < 2; ++mt) {
#pragma unroll
    for (int j = 0; j < 4; ++j) {
      const int cl = ngrp * 64 + j * 16 + (lane & 15);
      const float bv = Vb[colbase + cl];
#pragma unroll
      for (int r = 0; r < 8; ++r)
        tile[mgrp * 32 + mt * 16 + r + roff][cl] = acc[mt][j][r] + bv;
    }
  }
  __syncthreads();

  // per-row online (max, sum-exp): 4 threads/row, 64 cols each
  const int row = threadIdx.x >> 2;
  const int sub = threadIdx.x & 3;
  float m = -1e30f, s = 0.f;
  for (int c = sub * 64; c < sub * 64 + 64; ++c) {
    const float v = tile[row][c];
    if (v > m) { s = s * __expf(m - v) + 1.f; m = v; }
    else         s += __expf(v - m);
  }
  pmax[row][sub] = m; psum[row][sub] = s;
  __syncthreads();
  if (sub == 0) {
    float M = pmax[row][0];
#pragma unroll
    for (int i = 1; i < 4; ++i) M = fmaxf(M, pmax[row][i]);
    float S = 0.f;
#pragma unroll
    for (int i = 0; i < 4; ++i) S += psum[row][i] * __expf(pmax[row][i] - M);
    part[(size_t)(rowbase + row) * kParts + blockIdx.y] = make_float2(M, S);
  }
}

// ------- per-row: merge 125 partials + fp32 target logit -> nll -------
__global__ __launch_bounds__(128) void row_nll(const float2* __restrict__ part,
    const float* __restrict__ HSf, const float* __restrict__ VW,
    const float* __restrict__ Vb, const int* __restrict__ y,
    float* __restrict__ nll) {
  __shared__ float sm[128], ss[128], sd[128];
  const int r = blockIdx.x;
  const int t = threadIdx.x;
  float m = -1e30f, s = 0.f;
  if (t < kParts) { const float2 p = part[(size_t)r * kParts + t]; m = p.x; s = p.y; }
  const int yv = y[r];
  float d = 0.f;
  for (int j = t; j < kH; j += 128)
    d += HSf[(size_t)r * kH + j] * VW[(size_t)yv * kH + j];
  sm[t] = m; ss[t] = s; sd[t] = d;
  __syncthreads();
  for (int off = 64; off > 0; off >>= 1) {
    if (t < off) {
      const float m2 = sm[t + off], s2 = ss[t + off];
      const float M = fmaxf(sm[t], m2);
      ss[t] = ss[t] * __expf(sm[t] - M) + s2 * __expf(m2 - M);
      sm[t] = M;
      sd[t] += sd[t + off];
    }
    __syncthreads();
  }
  if (t == 0) nll[r] = (sm[0] + __logf(ss[0])) - (sd[0] + Vb[yv]);
}

// ------- deterministic mean over 4096 rows -> out[0] -------
__global__ __launch_bounds__(256) void final_loss(const float* __restrict__ nll,
                                                  float* __restrict__ out) {
  __shared__ float s[256];
  float a = 0.f;
  for (int i = threadIdx.x; i < kR; i += 256) a += nll[i];
  s[threadIdx.x] = a;
  __syncthreads();
  for (int off = 128; off > 0; off >>= 1) {
    if (threadIdx.x < off) s[threadIdx.x] += s[threadIdx.x + off];
    __syncthreads();
  }
  if (threadIdx.x == 0) out[0] = s[0] * (1.0f / (float)kR);
}

// ------- h_n = hs[:, S-1, :] -> out[1..] -------
__global__ __launch_bounds__(256) void copy_hn(const float* __restrict__ HSf,
                                               float* __restrict__ out) {
  const int i = blockIdx.x * 256 + threadIdx.x;
  const int b = i >> 10, h = i & 1023;
  out[1 + i] = HSf[((size_t)b * kS + (kS - 1)) * kH + h];
}

extern "C" void kernel_launch(void* const* d_in, const int* in_sizes, int n_in,
                              void* d_out, int out_size, void* d_ws, size_t ws_size,
                              hipStream_t stream) {
  (void)in_sizes; (void)n_in; (void)out_size; (void)ws_size;
  const float* x   = (const float*)d_in[0];
  const int*   y   = (const int*)d_in[1];
  const float* h0  = (const float*)d_in[2];
  const float* Wih = (const float*)d_in[3];
  const float* Whh = (const float*)d_in[4];
  const float* bih = (const float*)d_in[5];
  const float* bhh = (const float*)d_in[6];
  const float* Vw  = (const float*)d_in[7];
  const float* Vb  = (const float*)d_in[8];
  float* out = (float*)d_out;

  size_t off = 0;
  auto take = [&](size_t bytes) -> void* {
    void* p = (char*)d_ws + off;
    off += (bytes + 255) & ~(size_t)255;
    return p;
  };
  __bf16* xbf  = (__bf16*)take((size_t)kR * kE * 2);
  __bf16* wihb = (__bf16*)take((size_t)kH * kE * 2);
  __bf16* whhb = (__bf16*)take((size_t)kH * kH * 2);
  __bf16* vwb  = (__bf16*)take((size_t)kV * kH * 2);
  float*  xp   = (float*)take((size_t)kR * kH * 4);
  float*  hsf  = (float*)take((size_t)kR * kH * 4);
  __bf16* hsb  = (__bf16*)take((size_t)kR * kH * 2);
  __bf16* h0b  = (__bf16*)take((size_t)kB * kH * 2);
  __bf16* h1b  = (__bf16*)take((size_t)kB * kH * 2);
  float2* part = (float2*)take((size_t)kR * kParts * 8);
  float*  nll  = (float*)take((size_t)kR * 4);

  auto cvt = [&](const float* s, __bf16* dd, int n) {
    cvt_bf16<<<dim3((n / 4 + 255) / 256), dim3(256), 0, stream>>>(s, dd, n);
  };
  cvt(x,   xbf,  kR * kE);
  cvt(Wih, wihb, kH * kE);
  cvt(Whh, whhb, kH * kH);
  cvt(Vw,  vwb,  kV * kH);
  cvt(h0,  h0b,  kB * kH);

  gemm_xp<<<dim3(kR / 16, kH / 64), dim3(128), 0, stream>>>(xbf, wihb, bih, bhh, xp);

  const __bf16* hsrc = h0b;
  __bf16* hdst = h1b;
  for (int t = 0; t < kS; ++t) {
    rnn_step<<<dim3(kH / 16), dim3(128), 0, stream>>>(hsrc, whhb, xp, t, hsf, hsb, hdst);
    __bf16* tmp = (__bf16*)hsrc; hsrc = hdst; hdst = tmp;
  }

  vocab_lse<<<dim3(kR / 64, kParts), dim3(256), 0, stream>>>(hsb, vwb, Vb, part);
  row_nll<<<dim3(kR), dim3(128), 0, stream>>>(part, hsf, Vw, Vb, y, nll);
  final_loss<<<dim3(1), dim3(256), 0, stream>>>(nll, out);
  copy_hn<<<dim3((kB * kH) / 256), dim3(256), 0, stream>>>(hsf, out);
}
